// _Float8MM_13675175871041
// MI455X (gfx1250) — 8-bit matrix operand order repaired and checked by reading, compiled, not run on hardware
//
#include <hip/hip_runtime.h>

#define FP8_MAX 448.0f
#define K_DIM 256

typedef int   v16i_t __attribute__((ext_vector_type(16)));
typedef float v8f_t  __attribute__((ext_vector_type(8)));
typedef float v4f_t  __attribute__((ext_vector_type(4)));

__device__ __forceinline__ unsigned int f32_to_e4m3(float x) {
  unsigned int u    = __float_as_uint(x);
  unsigned int sign = (u >> 24) & 0x80u;
  unsigned int ax   = u & 0x7FFFFFFFu;
  if (ax >= 0x3C800000u) {
    unsigned int lsb = (ax >> 20) & 1u;
    unsigned int r   = ax + 0x0007FFFFu + lsb;
    unsigned int e   = (r >> 23) - 120u;
    unsigned int m   = (r >> 20) & 7u;
    if (e > 15u) { e = 15u; m = 6u; }
    return sign | (e << 4) | m;
  } else {
    float af = __uint_as_float(ax);
    unsigned int m = (unsigned int)rintf(af * 512.0f);
    return sign | m;
  }
}

__device__ __forceinline__ float clipq(float v, float s) {
  return fminf(fmaxf(v * s, -FP8_MAX), FP8_MAX);
}

__global__ void k_init(unsigned int* amax) {
  if (threadIdx.x < 2) amax[threadIdx.x] = 0u;
}

__global__ void k_amax(const float* __restrict__ x, int n,
                       unsigned int* __restrict__ out) {
  __shared__ float warpmax[8];
  float m = 0.0f;
  for (int i = blockIdx.x * blockDim.x + threadIdx.x; i < n;
       i += blockDim.x * gridDim.x)
    m = fmaxf(m, fabsf(x[i]));
#pragma unroll
  for (int off = 16; off > 0; off >>= 1)
    m = fmaxf(m, __shfl_down(m, off, 32));
  if ((threadIdx.x & 31) == 0) warpmax[threadIdx.x >> 5] = m;
  __syncthreads();
  if (threadIdx.x == 0) {
    float t = warpmax[0];
#pragma unroll
    for (int w = 1; w < 8; ++w) t = fmaxf(t, warpmax[w]);
    atomicMax(out, __float_as_uint(t));
  }
}

__global__ void k_quant_a(const float* __restrict__ A,
                          unsigned char* __restrict__ Aq,
                          const unsigned int* __restrict__ amax, int n4) {
  const float am = fmaxf(__uint_as_float(amax[0]), 1e-12f);
  const float s  = FP8_MAX / am;
  int i = blockIdx.x * blockDim.x + threadIdx.x;
  if (i >= n4) return;
  v4f_t v = __builtin_nontemporal_load(reinterpret_cast<const v4f_t*>(A) + i);
  unsigned int p = f32_to_e4m3(clipq(v[0], s))
                 | (f32_to_e4m3(clipq(v[1], s)) << 8)
                 | (f32_to_e4m3(clipq(v[2], s)) << 16)
                 | (f32_to_e4m3(clipq(v[3], s)) << 24);
  reinterpret_cast<unsigned int*>(Aq)[i] = p;
}

__global__ void k_quant_b(const float* __restrict__ B,
                          unsigned char* __restrict__ Bq,
                          const unsigned int* __restrict__ amax, int N) {
  const float am = fmaxf(__uint_as_float(amax[1]), 1e-12f);
  const float s  = FP8_MAX / am;
  int i = blockIdx.x * blockDim.x + threadIdx.x;
  if (i >= N * (K_DIM / 4)) return;
  int n  = i & (N - 1);
  int k4 = i / N;
  int k  = k4 * 4;
  unsigned int p = 0;
#pragma unroll
  for (int j = 0; j < 4; ++j) {
    float v = __builtin_nontemporal_load(B + (size_t)(k + j) * N + n);
    p |= f32_to_e4m3(clipq(v, s)) << (8 * j);
  }
  *reinterpret_cast<unsigned int*>(Bq + (size_t)n * K_DIM + k) = p;
}

__global__ __launch_bounds__(256) void k_gemm(
    const unsigned char* __restrict__ Aq, const unsigned char* __restrict__ Bq,
    const unsigned int* __restrict__ amax, float* __restrict__ C,
    int M, int N) {
  const int lane = threadIdx.x & 31;
  const int wave = threadIdx.x >> 5;
  const int half = lane >> 4;
  const int lx   = lane & 15;

  const int wm = blockIdx.y * 64  + (wave >> 2) * 32;
  const int wn = blockIdx.x * 128 + (wave & 3) * 32;

  v16i_t a[2][2];
#pragma unroll
  for (int mt = 0; mt < 2; ++mt) {
    const unsigned char* arow =
        Aq + (size_t)(wm + mt * 16 + lx) * K_DIM + half * 16;
#pragma unroll
    for (int kt = 0; kt < 2; ++kt) {
      uint4* ap = reinterpret_cast<uint4*>(&a[mt][kt]);
#pragma unroll
      for (int t = 0; t < 4; ++t)
        ap[t] = *reinterpret_cast<const uint4*>(arow + kt * 128 + t * 32);
    }
  }

  v8f_t acc[2][2];
#pragma unroll
  for (int mt = 0; mt < 2; ++mt)
#pragma unroll
    for (int nt = 0; nt < 2; ++nt) acc[mt][nt] = (v8f_t)0.0f;

#pragma unroll
  for (int nt = 0; nt < 2; ++nt) {
    v16i_t b[2];
    const unsigned char* bcol =
        Bq + (size_t)(wn + nt * 16 + lx) * K_DIM + half * 16;
#pragma unroll
    for (int kt = 0; kt < 2; ++kt) {
      uint4* bp = reinterpret_cast<uint4*>(&b[kt]);
#pragma unroll
      for (int g = 0; g < 4; ++g)
        bp[g] = *reinterpret_cast<const uint4*>(bcol + kt * 128 + g * 32);
    }
#pragma unroll
    for (int mt = 0; mt < 2; ++mt) {
      acc[mt][nt] = __builtin_amdgcn_wmma_f32_16x16x128_fp8_fp8(
          a[mt][0], b[0], (short)0, acc[mt][nt], false, false);
      acc[mt][nt] = __builtin_amdgcn_wmma_f32_16x16x128_fp8_fp8(
          a[mt][1], b[1], (short)0, acc[mt][nt], false, false);
    }
  }

  const float ama = fmaxf(__uint_as_float(amax[0]), 1e-12f);
  const float amb = fmaxf(__uint_as_float(amax[1]), 1e-12f);
  const float d   = (ama * amb) * (1.0f / (FP8_MAX * FP8_MAX));

#pragma unroll
  for (int mt = 0; mt < 2; ++mt)
#pragma unroll
    for (int nt = 0; nt < 2; ++nt) {
      float* crow =
          C + (size_t)(wm + mt * 16 + half * 8) * N + (wn + nt * 16 + lx);
#pragma unroll
      for (int r = 0; r < 8; ++r)
        __builtin_nontemporal_store(acc[mt][nt][r] * d, crow + (size_t)r * N);
    }
}

extern "C" void kernel_launch(void* const* d_in, const int* in_sizes, int n_in,
                              void* d_out, int out_size, void* d_ws,
                              size_t ws_size, hipStream_t stream) {
  const float* A = (const float*)d_in[0];
  const float* B = (const float*)d_in[1];
  float*       C = (float*)d_out;

  const int K = K_DIM;
  const int M = in_sizes[0] / K;
  const int N = in_sizes[1] / K;

  unsigned int*  amax = (unsigned int*)d_ws;
  unsigned char* Aq   = (unsigned char*)d_ws + 256;
  unsigned char* Bq   = Aq + (size_t)M * K;

  k_init<<<1, 32, 0, stream>>>(amax);
  k_amax<<<1024, 256, 0, stream>>>(A, M * K, amax + 0);
  k_amax<<<1024, 256, 0, stream>>>(B, N * K, amax + 1);
  k_quant_a<<<(M * K / 4 + 255) / 256, 256, 0, stream>>>(A, Aq, amax, M * K / 4);
  k_quant_b<<<(N * K / 4 + 255) / 256, 256, 0, stream>>>(B, Bq, amax, N);

  dim3 grid(N / 128, M / 64);
  k_gemm<<<grid, 256, 0, stream>>>(Aq, Bq, amax, C, M, N);
}
